// SPIntraAttModuleV2_87505663688957
// MI455X (gfx1250) — matrix operand order repaired and checked by reading, not compiled, not run on hardware
//
#include <hip/hip_runtime.h>
#include <hip/hip_bf16.h>

// ---------------- problem constants ----------------
#define B_    2
#define C_    256
#define HW_   16384          // 128*128
#define NSP_  256
#define TOPK_ 128
#define QK_   256
#define NPROJ 768            // q(256) + k(256) + v(256)
#define HEADS_ 8
#define HD_   32             // head dim
#define SCALE_ 0.17677669529663687f   // 32^-0.5

typedef _Float16 v8h  __attribute__((ext_vector_type(8)));
typedef _Float16 v16h __attribute__((ext_vector_type(16)));
typedef float    v8f  __attribute__((ext_vector_type(8)));
typedef unsigned int u32x4 __attribute__((ext_vector_type(4)));
typedef int          i32x4 __attribute__((ext_vector_type(4)));
typedef int          i32x8 __attribute__((ext_vector_type(8)));

#if defined(__has_builtin)
#if __has_builtin(__builtin_amdgcn_tensor_load_to_lds)
#define HAVE_TDM 1
#endif
#endif
#ifndef HAVE_TDM
#define HAVE_TDM 0
#endif

static __device__ inline v8f wmma_f16(v16h a, v16h b, v8f c) {
  // D = A(16x32 f16) * B(32x16 f16) + C(16x16 f32)
  return __builtin_amdgcn_wmma_f32_16x16x32_f16(false, a, false, b, (short)0, c,
                                                false, false);
}

static __device__ inline v8f zero8() {
  v8f z = {0.f,0.f,0.f,0.f,0.f,0.f,0.f,0.f};
  return z;
}

// A fragment, contiguous k order: lane L (g = L>>4) holds row m = L&15,
// K in [g*16, g*16+16). `row` already offset by k0.
static __device__ inline v16h load_a_frag(const _Float16* row, int g) {
  v8h lo = *(const v8h*)(row + g * 16);
  v8h hi = *(const v8h*)(row + g * 16 + 8);
  v16h r;
#pragma unroll
  for (int i = 0; i < 8; i++) { r[i] = lo[i]; r[i + 8] = hi[i]; }
  return r;
}

#if HAVE_TDM
// Issue a TDM 2D f16 tile load: tile (tile_d1 rows x tile_d0 elems) from a
// row-major tensor with row stride `stride0` (elements), into contiguous LDS.
// D# layout per CDNA5 ISA ch.8 (groups 0/1; groups 2/3 zero for 2D tensors).
static __device__ inline void tdm_load_2d_f16(unsigned lds_addr,
                                              unsigned long long gaddr,
                                              unsigned tensor_d0,
                                              unsigned tensor_d1,
                                              unsigned tile_d0,
                                              unsigned tile_d1,
                                              unsigned long long stride0) {
  u32x4 g0;
  g0[0] = 1u;                                        // count=1, user D#
  g0[1] = lds_addr;                                  // LDS byte address
  g0[2] = (unsigned)(gaddr & 0xFFFFFFFFull);         // global_addr[31:0]
  g0[3] = (unsigned)((gaddr >> 32) & 0x1FFFFFFull)   // global_addr[56:32]
          | (2u << 30);                              // type = 2 ("image")

  i32x8 g1;
  g1[0] = (int)(1u << 16);                           // data_size=1 (2B), mask=0
  g1[1] = (int)((tensor_d0 & 0xFFFFu) << 16);        // tensor_dim0[15:0]
  g1[2] = (int)(((tensor_d0 >> 16) & 0xFFFFu) |
                ((tensor_d1 & 0xFFFFu) << 16));      // td0[31:16] | td1[15:0]
  g1[3] = (int)(((tensor_d1 >> 16) & 0xFFFFu) |
                ((tile_d0 & 0xFFFFu) << 16));        // td1[31:16] | tile_d0
  g1[4] = (int)(tile_d1 & 0xFFFFu);                  // tile_d1 | tile_d2=0
  g1[5] = (int)(unsigned)(stride0 & 0xFFFFFFFFull);  // dim0_stride[31:0]
  g1[6] = (int)(unsigned)((stride0 >> 32) & 0xFFFFull); // dim0_stride[47:32]
  g1[7] = 0;                                         // dim1_stride = 0

  i32x4 gz = {0, 0, 0, 0};
#if __clang_major__ >= 23
  i32x8 gz8 = {0, 0, 0, 0, 0, 0, 0, 0};
  __builtin_amdgcn_tensor_load_to_lds(g0, g1, gz, gz, gz8, 0);
#else
  __builtin_amdgcn_tensor_load_to_lds(g0, g1, gz, gz, 0);
#endif
}
#endif

// ---------------- kernel: pack weights to f16 [wq;wk;wv] (768x256) ----------------
__global__ __launch_bounds__(256) void pack_w(const float* __restrict__ wq,
                                              const float* __restrict__ wk,
                                              const float* __restrict__ wv,
                                              _Float16* __restrict__ wcat) {
  int i = blockIdx.x * 256 + threadIdx.x;           // 0 .. 768*256-1
  int n = i >> 8;
  float v = (n < 256) ? wq[i] : (n < 512 ? wk[i - 256 * 256] : wv[i - 512 * 256]);
  wcat[i] = (_Float16)v;
}

// ---------------- kernel: LayerNorm over C, write xh (B*HW, C) f16 ----------------
__global__ __launch_bounds__(256) void ln_kernel(const float* __restrict__ x,
                                                 const float* __restrict__ gamma,
                                                 const float* __restrict__ beta,
                                                 _Float16* __restrict__ xh) {
  int b = blockIdx.y;
  int p0 = blockIdx.x * 256;
  int tid = threadIdx.x;
  const float* xb = x + (size_t)b * C_ * HW_ + p0 + tid;   // stride HW_ over c
  float s = 0.f, ss = 0.f;
  for (int c = 0; c < C_; c++) {
    float v = xb[(size_t)c * HW_];
    s += v; ss += v * v;
  }
  float mu   = s * (1.f / C_);
  float var  = ss * (1.f / C_) - mu * mu;
  float rstd = rsqrtf(var + 1e-6f);

  __shared__ _Float16 T[256][64];
  for (int c0 = 0; c0 < C_; c0 += 64) {
    for (int cc = 0; cc < 64; cc++) {
      int c = c0 + cc;
      float v = xb[(size_t)c * HW_];
      T[tid][cc] = (_Float16)((v - mu) * rstd * gamma[c] + beta[c]);
    }
    __syncthreads();
    size_t base = ((size_t)b * HW_ + p0) * C_ + c0;
    for (int i = 0; i < 64; i++) {
      int flat = i * 256 + tid;
      int pl = flat >> 6, cl = flat & 63;
      xh[base + (size_t)pl * C_ + cl] = T[pl][cl];
    }
    __syncthreads();
  }
}

// ---------------- kernel: QKV projection GEMM (M=32768, N=768, K=256, NT) ------
// Block computes a 128x64 tile; A/B k-slices staged into LDS by the Tensor Data
// Mover (one issue per block, synchronized via TENSORcnt + barrier). Each wave
// computes 16 rows x 64 cols via 4 WMMA accumulators per k-step.
__global__ __launch_bounds__(256) void gemm_qkv(const _Float16* __restrict__ xh,
                                                const _Float16* __restrict__ wcat,
                                                _Float16* __restrict__ qkv,
                                                float* __restrict__ v32) {
  __shared__ _Float16 As[128 * 32];   // 8 KiB: rows of xh, 32 halves each
  __shared__ _Float16 Bs[64 * 32];    // 4 KiB: rows of wcat

  int tid = threadIdx.x, lane = tid & 31, w = tid >> 5, g = lane >> 4;
  int m0 = blockIdx.x * 128;          // 0..32640
  int n0 = blockIdx.y * 64;           // 0..704

  v8f acc[4] = {zero8(), zero8(), zero8(), zero8()};

  for (int k0 = 0; k0 < QK_; k0 += 32) {
#if HAVE_TDM
    if (tid == 0) {
      tdm_load_2d_f16((unsigned)(size_t)(void*)As,
                      (unsigned long long)(size_t)(xh + (size_t)m0 * QK_ + k0),
                      QK_, B_ * HW_, 32, 128, QK_);
      tdm_load_2d_f16((unsigned)(size_t)(void*)Bs,
                      (unsigned long long)(size_t)(wcat + (size_t)n0 * QK_ + k0),
                      QK_, NPROJ, 32, 64, QK_);
      __builtin_amdgcn_s_wait_tensorcnt(0);
    }
#else
    {
      int r = tid >> 1, part = (tid & 1) * 16;              // 128 rows of A
      *(v16h*)(As + r * 32 + part) =
          *(const v16h*)(xh + (size_t)(m0 + r) * QK_ + k0 + part);
      if (tid < 128)
        *(v16h*)(Bs + r * 32 + part) =
            *(const v16h*)(wcat + (size_t)(n0 + r) * QK_ + k0 + part);
    }
#endif
    __syncthreads();

    v16h a = load_a_frag(As + (w * 16 + (lane & 15)) * 32, g);
#pragma unroll
    for (int t = 0; t < 4; t++) {
      v16h bf = *(const v16h*)(Bs + (t * 16 + (lane & 15)) * 32 + g * 16);
      acc[t] = wmma_f16(a, bf, acc[t]);
    }
    __syncthreads();
  }

#pragma unroll
  for (int t = 0; t < 4; t++) {
#pragma unroll
    for (int r = 0; r < 8; r++) {
      int m = m0 + w * 16 + r + g * 8;
      int n = n0 + t * 16 + (lane & 15);
      float v = acc[t][r];
      qkv[(size_t)m * NPROJ + n] = (_Float16)v;
      if (n >= 512) v32[(size_t)m * C_ + (n - 512)] = v;   // keep V in f32
    }
  }
}

// ---------------- kernel: top-128 per affinity row (512 rows x 16384) ----------
__global__ __launch_bounds__(1024) void topk_kernel(const float* __restrict__ aff,
                                                    int* __restrict__ idx_o,
                                                    float* __restrict__ sims_o) {
  int rowb = blockIdx.x;                       // b*NSP + sp
  const float* arow = aff + (size_t)rowb * HW_;
  int tid = threadIdx.x;
  float vals[16];
#pragma unroll
  for (int j = 0; j < 16; j++) vals[j] = arow[j * 1024 + tid];

  __shared__ float wv_s[32];
  __shared__ int   wi_s[32];
  __shared__ int   bi_s;
  for (int it = 0; it < TOPK_; it++) {
    float bv = -3.4e38f; int bj = 0;
#pragma unroll
    for (int j = 0; j < 16; j++)
      if (vals[j] > bv) { bv = vals[j]; bj = j; }
    int gi = bj * 1024 + tid;
#pragma unroll
    for (int off = 16; off; off >>= 1) {
      float ov = __shfl_xor(bv, off, 32);
      int   oi = __shfl_xor(gi, off, 32);
      if (ov > bv) { bv = ov; gi = oi; }
    }
    if ((tid & 31) == 0) { wv_s[tid >> 5] = bv; wi_s[tid >> 5] = gi; }
    __syncthreads();
    if (tid < 32) {
      bv = wv_s[tid]; gi = wi_s[tid];
#pragma unroll
      for (int off = 16; off; off >>= 1) {
        float ov = __shfl_xor(bv, off, 32);
        int   oi = __shfl_xor(gi, off, 32);
        if (ov > bv) { bv = ov; gi = oi; }
      }
      if (tid == 0) {
        bi_s = gi;
        idx_o[(size_t)rowb * TOPK_ + it]  = gi;
        sims_o[(size_t)rowb * TOPK_ + it] = bv;
      }
    }
    __syncthreads();
    int win = bi_s;
    if (tid == (win & 1023)) vals[win >> 10] = -3.4e38f;
  }
}

// ---------------- kernel: out base = v^T (tiled transpose) ---------------------
__global__ __launch_bounds__(256) void init_out(const float* __restrict__ v32,
                                                float* __restrict__ out) {
  __shared__ float T[32][33];
  int b = blockIdx.z, p0 = blockIdx.x * 32, c0 = blockIdx.y * 32;
  int tx = threadIdx.x, ty = threadIdx.y;          // (32, 8)
  for (int r = 0; r < 32; r += 8)
    T[ty + r][tx] = v32[((size_t)b * HW_ + p0 + ty + r) * C_ + c0 + tx];
  __syncthreads();
  for (int r = 0; r < 32; r += 8)
    out[(size_t)b * C_ * HW_ + (size_t)(c0 + ty + r) * HW_ + p0 + tx] = T[tx][ty + r];
}

// ---------------- kernel: per-superpixel attention + scatter -------------------
// One block per (sp, b); 256 threads = 8 waves; loops over 8 heads.
__global__ __launch_bounds__(256) void attn_kernel(const _Float16* __restrict__ qkv,
                                                   const float* __restrict__ v32,
                                                   const int* __restrict__ idx,
                                                   const float* __restrict__ sims,
                                                   float* __restrict__ out) {
  extern __shared__ __align__(128) char smem_raw[];
  int*      idx_l  = (int*)smem_raw;                                   // 512 B
  float*    sims_l = (float*)(smem_raw + 512);                         // 512 B
  _Float16* qs     = (_Float16*)(smem_raw + 1024);                     // 128x32 f16
  _Float16* ks     = (_Float16*)(smem_raw + 1024 + 8192);              // 128x32 f16
  _Float16* vt     = (_Float16*)(smem_raw + 1024 + 16384);             // 32x128 f16 (c-major)
  float*    S      = (float*)(smem_raw + 1024 + 24576);                // 128x128 f32
  _Float16* P      = (_Float16*)(smem_raw + 1024 + 24576 + 65536);     // 128x128 f16

  int sp = blockIdx.x, b = blockIdx.y;
  int tid = threadIdx.x, lane = tid & 31, w = tid >> 5, g = lane >> 4;
  int row = b * NSP_ + sp;

  if (tid < TOPK_) {
    idx_l[tid]  = idx[(size_t)row * TOPK_ + tid];
    sims_l[tid] = sims[(size_t)row * TOPK_ + tid];
  }
  __syncthreads();

  for (int h = 0; h < HEADS_; h++) {
    // ---- gather Q, K (f16) and sims-scaled V^T (f32 -> f16) into LDS ----
    {
      int t = tid >> 1, part = (tid & 1) * 16;
      size_t base = ((size_t)b * HW_ + idx_l[t]) * NPROJ;
      v16h qv = *(const v16h*)(qkv + base + h * HD_ + part);
      v16h kv = *(const v16h*)(qkv + base + QK_ + h * HD_ + part);
      *(v16h*)(qs + t * HD_ + part) = qv;
      *(v16h*)(ks + t * HD_ + part) = kv;
      const float* vrow = v32 + ((size_t)b * HW_ + idx_l[t]) * C_ + h * HD_ + part;
      float sw = sims_l[t];
#pragma unroll
      for (int c = 0; c < 16; c++)
        vt[(part + c) * TOPK_ + t] = (_Float16)(sw * vrow[c]);
    }
    __syncthreads();

    // ---- S = Q * K^T  (M=128, N=128, K=32; wave w -> rows 16w..16w+15) ----
    {
      v16h a = load_a_frag(qs + (w * 16 + (lane & 15)) * HD_, g);
      int mrow = w * 16 + g * 8;
#pragma unroll
      for (int nt = 0; nt < 8; nt++) {
        v16h bf = *(const v16h*)(ks + (nt * 16 + (lane & 15)) * HD_ + g * 16);
        v8f acc = wmma_f16(a, bf, zero8());
        int ncol = nt * 16 + (lane & 15);
#pragma unroll
        for (int r = 0; r < 8; r++) S[(mrow + r) * TOPK_ + ncol] = acc[r];
      }
    }
    __syncthreads();

    // ---- softmax rows (scale folded in), P = f16(softmax(S*scale)) ----
    if (tid < TOPK_) {
      float* srow = S + tid * TOPK_;
      float mx = -3.4e38f;
      for (int j = 0; j < TOPK_; j++) mx = fmaxf(mx, srow[j]);
      const float kf = SCALE_ * 1.4426950408889634f;   // scale * log2(e)
      float sum = 0.f;
      for (int j = 0; j < TOPK_; j++) {
        float e = exp2f((srow[j] - mx) * kf);
        srow[j] = e; sum += e;
      }
      float inv = 1.f / sum;
      _Float16* prow = P + tid * TOPK_;
      for (int j = 0; j < TOPK_; j++) prow[j] = (_Float16)(srow[j] * inv);
    }
    __syncthreads();

    // ---- O = P * (sims*V)  (M=128, N=32, K=128), scatter-add with sims[t] ----
#pragma unroll
    for (int nt = 0; nt < 2; nt++) {
      v8f acc = zero8();
#pragma unroll
      for (int k0 = 0; k0 < TOPK_; k0 += 32) {
        v16h a  = load_a_frag(P + (w * 16 + (lane & 15)) * TOPK_ + k0, g);
        v16h bf = *(const v16h*)(vt + (nt * 16 + (lane & 15)) * TOPK_ + k0 + g * 16);
        acc = wmma_f16(a, bf, acc);
      }
      int c = h * HD_ + nt * 16 + (lane & 15);
      float* obase = out + ((size_t)b * C_ + c) * HW_;
#pragma unroll
      for (int r = 0; r < 8; r++) {
        int t = w * 16 + r + g * 8;
        atomicAdd(obase + idx_l[t], sims_l[t] * acc[r]);
      }
    }
    __syncthreads();
  }
}

// ---------------- launch -------------------------------------------------------
extern "C" void kernel_launch(void* const* d_in, const int* in_sizes, int n_in,
                              void* d_out, int out_size, void* d_ws, size_t ws_size,
                              hipStream_t stream) {
  (void)in_sizes; (void)n_in; (void)out_size; (void)ws_size;
  const float* x     = (const float*)d_in[0];
  const float* aff   = (const float*)d_in[1];
  const float* gamma = (const float*)d_in[2];
  const float* beta  = (const float*)d_in[3];
  const float* wq    = (const float*)d_in[4];
  const float* wk    = (const float*)d_in[5];
  const float* wv    = (const float*)d_in[6];
  float* out = (float*)d_out;

  // workspace carve (all offsets 128B aligned)
  char* ws = (char*)d_ws;
  _Float16* xh   = (_Float16*)(ws);                               // 16 MiB
  _Float16* wcat = (_Float16*)(ws + 16777216);                    // 384 KiB
  _Float16* qkv  = (_Float16*)(ws + 16777216 + 393216);           // 48 MiB
  float*    v32  = (float*)   (ws + 16777216 + 393216 + 50331648);// 32 MiB
  int*      idxb = (int*)     (ws + 16777216 + 393216 + 50331648 + 33554432);
  float*    simb = (float*)   (ws + 16777216 + 393216 + 50331648 + 33554432 + 262144);

  pack_w<<<NPROJ, 256, 0, stream>>>(wq, wk, wv, wcat);
  ln_kernel<<<dim3(HW_ / 256, B_), 256, 0, stream>>>(x, gamma, beta, xh);
  gemm_qkv<<<dim3(32768 / 128, NPROJ / 64), 256, 0, stream>>>(xh, wcat, qkv, v32);
  topk_kernel<<<B_ * NSP_, 1024, 0, stream>>>(aff, idxb, simb);
  init_out<<<dim3(HW_ / 32, C_ / 32, B_), dim3(32, 8), 0, stream>>>(v32, out);
  size_t attn_lds = 1024 + 3 * 8192 + 65536 + 32768;   // 123,904 B
  attn_kernel<<<dim3(NSP_, B_), 256, attn_lds, stream>>>(qkv, v32, idxb, simb, out);
}
